// VectorMSE_66786741453356
// MI455X (gfx1250) — compile-verified, matrix operand order checked by reading, not run on hardware
//
#include <hip/hip_runtime.h>
#include <hip/hip_bf16.h>

// ---------------------------------------------------------------------------
// InfoNCE-with-Gaussian-logits loss, fused flash-softmax formulation.
//
//   s_ij  = (p_i . g_j - 0.5*||g_j||^2) / sigma^2        (row terms cancel)
//   loss  = (2*sigma^2 / B) * sum_i [ logsumexp_j(s_ij) - s_ii ]
//
// Matmul done with v_wmma_f32_16x16x32_bf16 using a 3-WMMA hi/lo split for
// ~fp32 accuracy. B=8192, D=256.
// ---------------------------------------------------------------------------

typedef __attribute__((ext_vector_type(16))) __bf16 v16bf;
typedef __attribute__((ext_vector_type(8)))  __bf16 v8bf;
typedef __attribute__((ext_vector_type(8)))  float  v8f;

#define BN 8192
#define DN 256
#define NSPLIT 8           // column splits per row-tile
#define COLS_PER_SPLIT (BN / NSPLIT)   // 1024

union PackBF {
    v16bf v;
    v8bf  h[2];
};

// ---------------------------------------------------------------------------
// Kernel 0: zero the output accumulator.
// ---------------------------------------------------------------------------
__global__ void nce_zero(float* out) {
    if (threadIdx.x == 0 && blockIdx.x == 0) out[0] = 0.0f;
}

// ---------------------------------------------------------------------------
// Kernel 1: convert pred/gt to bf16 hi/lo pairs; compute colBias[j]=-0.5*||g_j||^2.
// One wave per row; 8 waves per block; rows [0,8192)=pred, [8192,16384)=gt.
// ---------------------------------------------------------------------------
__global__ __launch_bounds__(256) void nce_prep(
    const float* __restrict__ pred, const float* __restrict__ gt,
    __bf16* __restrict__ p_hi, __bf16* __restrict__ p_lo,
    __bf16* __restrict__ g_hi, __bf16* __restrict__ g_lo,
    float* __restrict__ colBias)
{
    const int wave = threadIdx.x >> 5;
    const int lane = threadIdx.x & 31;
    const int row  = blockIdx.x * 8 + wave;          // 0 .. 16383
    const bool isG = row >= BN;
    const int r    = isG ? row - BN : row;

    const float* src = (isG ? gt : pred) + (size_t)r * DN;
    __bf16* dhi = (isG ? g_hi : p_hi) + (size_t)r * DN;
    __bf16* dlo = (isG ? g_lo : p_lo) + (size_t)r * DN;

    float ss = 0.0f;
    #pragma unroll
    for (int k = lane; k < DN; k += 32) {
        float x  = src[k];
        __bf16 h = (__bf16)x;
        float lo = x - (float)h;
        dhi[k] = h;
        dlo[k] = (__bf16)lo;
        ss += x * x;
    }
    // full-wave reduction (wave32)
    #pragma unroll
    for (int m = 1; m < 32; m <<= 1) ss += __shfl_xor(ss, m, 32);
    if (isG && lane == 0) colBias[r] = -0.5f * ss;
}

// ---------------------------------------------------------------------------
// Kernel 2: main fused GEMM + online softmax.
// Grid: 512 blocks x 256 threads = 4096 wave-tasks.
// task = rowTile*8 + split : wave computes s_{i,j} for 16 rows x 1024 cols,
// emitting per-(row,split) partial (max, sumExp) + the diagonal element.
// ---------------------------------------------------------------------------
__global__ __launch_bounds__(256) void nce_main(
    const __bf16* __restrict__ p_hi, const __bf16* __restrict__ p_lo,
    const __bf16* __restrict__ g_hi, const __bf16* __restrict__ g_lo,
    const float* __restrict__ colBias, const float* __restrict__ sigma,
    float* __restrict__ mxPart, float* __restrict__ sumPart,
    float* __restrict__ diagArr)
{
    const int wave  = threadIdx.x >> 5;
    const int lane  = threadIdx.x & 31;
    const int laneN = lane & 15;         // column-within-tile / row-within-tile
    const int hi    = lane >> 4;         // lane half selects K-interleave + row offset
    const int task  = blockIdx.x * 8 + wave;
    const int split   = task & (NSPLIT - 1);
    const int rowTile = task >> 3;       // 0..511 ; 8 waves of a block share it
    const int rowBase = rowTile * 16;

    const float sg    = sigma[0];
    const float nv    = sg * sg;
    const float invNv = 1.0f / nv;

    // ---- preload A (pred) tiles: 16 rows x 256 K, split into 8 k-tiles of 32.
    // ISA A-layout (16-bit 16x32): lanes 0-15 hold K {0..7,16..23},
    // lanes 16-31 hold K {8..15,24..31}; element e<8 -> K kbase+e,
    // element e>=8 -> K kbase+16+(e-8).  Two 16B loads per lane per tile.
    const int kbase = hi * 8;
    const __bf16* aRowHi = p_hi + (size_t)(rowBase + laneN) * DN + kbase;
    const __bf16* aRowLo = p_lo + (size_t)(rowBase + laneN) * DN + kbase;

    v16bf aHi[8], aLo[8];
    #pragma unroll
    for (int kt = 0; kt < 8; ++kt) {
        PackBF u;
        u.h[0] = *(const v8bf*)(aRowHi + kt * 32);
        u.h[1] = *(const v8bf*)(aRowHi + kt * 32 + 16);
        aHi[kt] = u.v;
        u.h[0] = *(const v8bf*)(aRowLo + kt * 32);
        u.h[1] = *(const v8bf*)(aRowLo + kt * 32 + 16);
        aLo[kt] = u.v;
    }

    // online-softmax state: 8 rows per lane (row m = v + 8*hi), column subset
    // j = j0 + laneN over the 64 column tiles of this split.
    float mx[8], sm[8], dg[8];
    #pragma unroll
    for (int v = 0; v < 8; ++v) { mx[v] = -3.0e38f; sm[v] = 0.0f; dg[v] = 0.0f; }

    const int j0beg = split * COLS_PER_SPLIT;
    const int j0end = j0beg + COLS_PER_SPLIT;

    for (int j0 = j0beg; j0 < j0end; j0 += 16) {
        const int gRow = j0 + laneN;     // B-matrix column n = g row (j0+n)
        const __bf16* gbHi = g_hi + (size_t)gRow * DN + kbase;
        const __bf16* gbLo = g_lo + (size_t)gRow * DN + kbase;

        // prefetch next column tile's g rows (global_prefetch_b8)
        if (j0 + 16 < j0end) {
            __builtin_prefetch(g_hi + (size_t)(gRow + 16) * DN, 0, 1);
            __builtin_prefetch(g_lo + (size_t)(gRow + 16) * DN, 0, 1);
        }

        v8f acc = {};
        #pragma unroll
        for (int kt = 0; kt < 8; ++kt) {
            PackBF ub;
            ub.h[0] = *(const v8bf*)(gbHi + kt * 32);
            ub.h[1] = *(const v8bf*)(gbHi + kt * 32 + 16);
            v16bf bHi = ub.v;
            ub.h[0] = *(const v8bf*)(gbLo + kt * 32);
            ub.h[1] = *(const v8bf*)(gbLo + kt * 32 + 16);
            v16bf bLo = ub.v;
            // fp32-accurate split product: hi*hi + hi*lo + lo*hi
            acc = __builtin_amdgcn_wmma_f32_16x16x32_bf16(
                false, aHi[kt], false, bHi, (short)0, acc, false, false);
            acc = __builtin_amdgcn_wmma_f32_16x16x32_bf16(
                false, aHi[kt], false, bLo, (short)0, acc, false, false);
            acc = __builtin_amdgcn_wmma_f32_16x16x32_bf16(
                false, aLo[kt], false, bHi, (short)0, acc, false, false);
        }

        const float cb = colBias[gRow];          // -0.5*||g_j||^2
        const bool diagTile = (j0 == rowBase);
        #pragma unroll
        for (int v = 0; v < 8; ++v) {
            float s  = (acc[v] + cb) * invNv;
            float mo = mx[v];
            float mn = fmaxf(mo, s);
            sm[v] = sm[v] * __expf(mo - mn) + __expf(s - mn);
            mx[v] = mn;
            // row = rowBase + v + 8*hi ; col = j0 + laneN ; equal iff laneN==v+8*hi
            if (diagTile && (laneN == v + kbase)) dg[v] = s;
        }
    }

    // butterfly-combine the 16 lanes sharing each row (halves stay separate:
    // lanes 0-15 own rows +0..7, lanes 16-31 own rows +8..15).
    #pragma unroll
    for (int m = 1; m < 16; m <<= 1) {
        #pragma unroll
        for (int v = 0; v < 8; ++v) {
            float omx = __shfl_xor(mx[v], m, 32);
            float osm = __shfl_xor(sm[v], m, 32);
            float odg = __shfl_xor(dg[v], m, 32);
            float nm  = fmaxf(mx[v], omx);
            sm[v] = sm[v] * __expf(mx[v] - nm) + osm * __expf(omx - nm);
            mx[v] = nm;
            dg[v] += odg;       // exactly one contributing lane, rest are 0
        }
    }

    if (laneN == 0) {           // lanes 0 and 16 write their 8 rows
        #pragma unroll
        for (int v = 0; v < 8; ++v) {
            int row = rowBase + v + kbase;
            mxPart[(size_t)row * NSPLIT + split]  = mx[v];
            sumPart[(size_t)row * NSPLIT + split] = sm[v];
            if ((rowBase / COLS_PER_SPLIT) == split) diagArr[row] = dg[v];
        }
    }
}

// ---------------------------------------------------------------------------
// Kernel 3: merge the 8 per-split partials per row, accumulate scaled loss.
// ---------------------------------------------------------------------------
__global__ __launch_bounds__(256) void nce_finish(
    const float* __restrict__ mxPart, const float* __restrict__ sumPart,
    const float* __restrict__ diagArr, const float* __restrict__ sigma,
    float* __restrict__ out)
{
    const int r    = blockIdx.x * 256 + threadIdx.x;   // row id, 32 blocks
    const int wave = threadIdx.x >> 5;
    const int lane = threadIdx.x & 31;

    float mx = -3.0e38f;
    #pragma unroll
    for (int s = 0; s < NSPLIT; ++s) mx = fmaxf(mx, mxPart[(size_t)r * NSPLIT + s]);
    float sm = 0.0f;
    #pragma unroll
    for (int s = 0; s < NSPLIT; ++s)
        sm += sumPart[(size_t)r * NSPLIT + s] * __expf(mxPart[(size_t)r * NSPLIT + s] - mx);

    float loss = (mx + __logf(sm)) - diagArr[r];       // lse_i - s_ii

    #pragma unroll
    for (int m = 1; m < 32; m <<= 1) loss += __shfl_xor(loss, m, 32);

    __shared__ float red[8];
    if (lane == 0) red[wave] = loss;
    __syncthreads();
    if (threadIdx.x == 0) {
        float t = 0.0f;
        #pragma unroll
        for (int i = 0; i < 8; ++i) t += red[i];
        float sg = sigma[0];
        float nv = sg * sg;
        atomicAdd(out, t * (2.0f * nv / (float)BN));
    }
}

// ---------------------------------------------------------------------------
extern "C" void kernel_launch(void* const* d_in, const int* in_sizes, int n_in,
                              void* d_out, int out_size, void* d_ws, size_t ws_size,
                              hipStream_t stream)
{
    const float* pred  = (const float*)d_in[0];
    const float* gt    = (const float*)d_in[1];
    const float* sigma = (const float*)d_in[2];
    float* out = (float*)d_out;

    // carve workspace (all offsets 256B-aligned)
    char* ws = (char*)d_ws;
    size_t off = 0;
    auto carve = [&](size_t bytes) -> void* {
        void* p = ws + off;
        off = (off + bytes + 255) & ~(size_t)255;
        return p;
    };
    const size_t matB = (size_t)BN * DN * sizeof(__bf16);   // 4 MB each
    __bf16* p_hi = (__bf16*)carve(matB);
    __bf16* p_lo = (__bf16*)carve(matB);
    __bf16* g_hi = (__bf16*)carve(matB);
    __bf16* g_lo = (__bf16*)carve(matB);
    float* colBias = (float*)carve((size_t)BN * sizeof(float));
    float* mxPart  = (float*)carve((size_t)BN * NSPLIT * sizeof(float));
    float* sumPart = (float*)carve((size_t)BN * NSPLIT * sizeof(float));
    float* diagArr = (float*)carve((size_t)BN * sizeof(float));

    nce_zero<<<1, 1, 0, stream>>>(out);
    nce_prep<<<(2 * BN) / 8, 256, 0, stream>>>(pred, gt, p_hi, p_lo, g_hi, g_lo, colBias);
    nce_main<<<(BN / 16) * NSPLIT / 8, 256, 0, stream>>>(
        p_hi, p_lo, g_hi, g_lo, colBias, sigma, mxPart, sumPart, diagArr);
    nce_finish<<<BN / 256, 256, 0, stream>>>(mxPart, sumPart, diagArr, sigma, out);
}
